// MultiHeadAttention_38457137168907
// MI455X (gfx1250) — matrix operand order repaired and checked by reading, not compiled, not run on hardware
//
#include <hip/hip_runtime.h>
#include <hip/hip_bf16.h>

// MI455X (gfx1250) multi-head attention, 12 layers, WMMA f32_16x16x32_f16 path.
// B=8 S=1024 D=768 H=12 HD=64. Compute-bound -> f16 WMMA, fp32 accum, fp32 x.

typedef __attribute__((ext_vector_type(16))) _Float16 v16h;
typedef __attribute__((ext_vector_type(8)))  _Float16 v8h;
typedef __attribute__((ext_vector_type(4)))  _Float16 v4h;
typedef __attribute__((ext_vector_type(8)))  float    v8f;
typedef __attribute__((ext_vector_type(4)))  float    v4f;

#define NB    8
#define SS    1024
#define DD    768
#define HH    12
#define HD    64

// ---- WMMA fragment helpers (layouts per CDNA5 ISA §7.12.2) ----
// A 16x32 f16: lane l -> row m=l&15, half=l>>4; v0..3 = K(8*half+0..7), v4..7 = K(16+8*half+0..7)
static __device__ __forceinline__ v16h ldA(const _Float16* rowp, int kb, int half) {
    union { v16h v; v8h h[2]; } u;
    u.h[0] = *(const v8h*)(rowp + kb + 8 * half);
    u.h[1] = *(const v8h*)(rowp + kb + 16 + 8 * half);
    return u.v;
}
// B 32x16 f16: lane l -> col n=l&15, half=l>>4; v0..3 = K(8*half+0..7), v4..7 = K(16+8*half+0..7) at fixed n
// colp points at element (K=0, n): data stored [n][K] with K contiguous.
static __device__ __forceinline__ v16h ldB(const _Float16* colp, int kb, int half) {
    union { v16h v; v8h h[2]; } u;
    u.h[0] = *(const v8h*)(colp + kb + 8 * half);
    u.h[1] = *(const v8h*)(colp + kb + 16 + 8 * half);
    return u.v;
}
static __device__ __forceinline__ v8f wmma32(v16h a, v16h b, v8f c) {
    return __builtin_amdgcn_wmma_f32_16x16x32_f16(false, a, false, b, (short)0, c, false, false);
}

// ---- weight prep: [H][K][64] f32 -> [H][64][K] f16 (transpose so B-frags are contiguous)
__global__ void cvt_wT_kernel(const float* __restrict__ W, _Float16* __restrict__ WT, int K) {
    int j = blockIdx.x * 256 + threadIdx.x;
    int total = HH * 64 * K;
    if (j >= total) return;
    int k = j % K;
    int n = (j / K) % 64;
    int h = j / (K * 64);
    WT[j] = (_Float16)W[((size_t)h * K + k) * 64 + n];
}

// vectorized fp32->f16 tile fill: 64 rows x 32 k  = 512 float4 groups over 384 threads
static __device__ __forceinline__ void fill_xtile(_Float16* dst, const float* __restrict__ xrow,
                                                  int kb, int tid) {
    for (int i = tid; i < 512; i += 384) {
        int rr = i >> 3, c4 = (i & 7) * 4;
        v4f f = *(const v4f*)(xrow + (size_t)rr * DD + kb + c4);
        v4h hv;
        hv[0] = (_Float16)f[0];
        hv[1] = (_Float16)f[1];
        hv[2] = (_Float16)f[2];
        hv[3] = (_Float16)f[3];
        *(v4h*)(dst + rr * 32 + c4) = hv;
    }
}

// ---- QKV projection: grid = B*H*(S/64), 384 thr = 12 waves
// wave w: matrix = w>>2 (0=q,1=k,2=v), m-strip = w&3 (16 rows of the 64-row s-tile)
__global__ __launch_bounds__(384) void qkv_proj_kernel(
    const float* __restrict__ x,
    const _Float16* __restrict__ WTq, const _Float16* __restrict__ WTk, const _Float16* __restrict__ WTv,
    const float* __restrict__ bq, const float* __restrict__ bk, const float* __restrict__ bv,
    _Float16* __restrict__ Qg, _Float16* __restrict__ Kg, _Float16* __restrict__ Vtg) {
    __shared__ __align__(16) _Float16 xs[2][64 * 32];  // double-buffered 4KB X tiles

    const int tid = threadIdx.x;
    const int wv = tid >> 5, lane = tid & 31;
    const int n16 = lane & 15, half = lane >> 4;
    const int stile = blockIdx.x & 15;
    const int bh = blockIdx.x >> 4;
    const int h = bh % HH;
    const int b = bh / HH;
    const int s0 = stile * 64;
    const int mat = wv >> 2, ms = wv & 3;

    const _Float16* WT = (mat == 0) ? WTq : (mat == 1) ? WTk : WTv;
    const float* bias = (mat == 0) ? bq : (mat == 1) ? bk : bv;
    const float* xrow = x + ((size_t)b * SS + s0) * DD;

    // column base pointers for the 4 N-tiles of this wave's output
    const _Float16* colp[4];
#pragma unroll
    for (int nt = 0; nt < 4; ++nt) colp[nt] = WT + (size_t)(h * 64 + nt * 16 + n16) * DD;

    v8f acc[4];
#pragma unroll
    for (int i = 0; i < 4; ++i)
#pragma unroll
        for (int r = 0; r < 8; ++r) acc[i][r] = 0.f;

    fill_xtile(xs[0], xrow, 0, tid);
    __syncthreads();

#pragma unroll 1
    for (int step = 0; step < DD / 32; ++step) {
        const int p = step & 1;
        const int kb = step * 32;
        // cluster all B-frag loads + A-frag load, then issue the 4 WMMAs back-to-back
        v16h bfr[4];
#pragma unroll
        for (int nt = 0; nt < 4; ++nt) bfr[nt] = ldB(colp[nt], kb, half);
        v16h a = ldA(&xs[p][(ms * 16 + n16) * 32], 0, half);
        if (step + 1 < DD / 32) fill_xtile(xs[p ^ 1], xrow, kb + 32, tid);  // overlap with WMMA
#pragma unroll
        for (int nt = 0; nt < 4; ++nt) acc[nt] = wmma32(a, bfr[nt], acc[nt]);
        __syncthreads();
    }

    const float qscale = (mat == 0) ? 0.125f : 1.0f;  // fold 1/sqrt(HD) into Q
#pragma unroll
    for (int nt = 0; nt < 4; ++nt) {
        const int n = nt * 16 + n16;
        const float bval = bias[h * 64 + n];
        if (mat == 2) {
            // V stored transposed [bh][hd][s]; 8 consecutive s per lane -> one 16B store
            v8h tv;
#pragma unroll
            for (int r = 0; r < 8; ++r) tv[r] = (_Float16)(acc[nt][r] + bval);
            _Float16* dst = Vtg + ((size_t)bh * 64 + n) * SS + s0 + ms * 16 + 8 * half;
            *(v8h*)dst = tv;
        } else {
            _Float16* dst = ((mat == 0) ? Qg : Kg) + ((size_t)bh * SS + s0 + ms * 16) * 64 + n;
#pragma unroll
            for (int r = 0; r < 8; ++r) {
                int m = r + 8 * half;
                dst[(size_t)m * 64] = (_Float16)((acc[nt][r] + bval) * qscale);
            }
        }
    }
}

// ---- flash attention + fused output projection: grid = B*H*(S/64), 128 thr = 4 waves
// wave w owns 16 query rows; online softmax state replicated across the 16-lane half.
__global__ __launch_bounds__(128) void attn_kernel(
    const _Float16* __restrict__ Qg, const _Float16* __restrict__ Kg, const _Float16* __restrict__ Vtg,
    const _Float16* __restrict__ WoT, const float* __restrict__ bo, float* __restrict__ xout) {
    __shared__ __align__(16) _Float16 plds[4][16][64];  // per-wave C->A relayout scratch

    const int tid = threadIdx.x;
    const int w = tid >> 5, lane = tid & 31;
    const int n16 = lane & 15, half = lane >> 4;
    const int stile = blockIdx.x & 15;
    const int bh = blockIdx.x >> 4;
    const int h = bh % HH;
    const int b = bh / HH;
    const int s0 = stile * 64;

    // Q A-fragments (rows s0+16w .. +15), kept in registers for all 16 t-tiles
    const _Float16* qrow = Qg + ((size_t)bh * SS + s0 + w * 16 + n16) * 64;
    const v16h qa0 = ldA(qrow, 0, half);
    const v16h qa1 = ldA(qrow, 32, half);

    const _Float16* Kbase = Kg + (size_t)bh * SS * 64;
    const _Float16* Vbase = Vtg + (size_t)bh * 64 * SS;

    float mst[8], lst[8];
    v8f acc[4];
#pragma unroll
    for (int r = 0; r < 8; ++r) { mst[r] = -3.0e38f; lst[r] = 0.f; }
#pragma unroll
    for (int nt = 0; nt < 4; ++nt)
#pragma unroll
        for (int r = 0; r < 8; ++r) acc[nt][r] = 0.f;

#pragma unroll 1
    for (int tb = 0; tb < SS; tb += 64) {
        if (tb + 64 < SS) {  // uniform branch: EXEC stays all-ones
            __builtin_prefetch(Kbase + (size_t)(tb + 64 + n16) * 64, 0, 3);
            __builtin_prefetch(Vbase + (size_t)n16 * SS + tb + 64, 0, 3);
        }
        // scores strip: 16 x 64 over this t-tile; cluster K-frag loads per sub-tile
        v8f sc[4];
#pragma unroll
        for (int tt = 0; tt < 4; ++tt) {
            const _Float16* kcol = Kbase + (size_t)(tb + tt * 16 + n16) * 64;
            v16h kf0 = ldB(kcol, 0, half);
            v16h kf1 = ldB(kcol, 32, half);
            v8f s;
#pragma unroll
            for (int r = 0; r < 8; ++r) s[r] = 0.f;
            s = wmma32(qa0, kf0, s);
            s = wmma32(qa1, kf1, s);
            sc[tt] = s;
        }
        // online softmax: row m = r + 8*half lives across the 16 lanes of this half
        float scl[8];
#pragma unroll
        for (int r = 0; r < 8; ++r) {
            float v = fmaxf(fmaxf(sc[0][r], sc[1][r]), fmaxf(sc[2][r], sc[3][r]));
            v = fmaxf(v, __shfl_xor(v, 1));
            v = fmaxf(v, __shfl_xor(v, 2));
            v = fmaxf(v, __shfl_xor(v, 4));
            v = fmaxf(v, __shfl_xor(v, 8));
            float mn = fmaxf(mst[r], v);
            scl[r] = __expf(mst[r] - mn);
            mst[r] = mn;
        }
#pragma unroll
        for (int tt = 0; tt < 4; ++tt)
#pragma unroll
            for (int r = 0; r < 8; ++r) sc[tt][r] = __expf(sc[tt][r] - mst[r]);
#pragma unroll
        for (int r = 0; r < 8; ++r) {
            float s = sc[0][r] + sc[1][r] + sc[2][r] + sc[3][r];
            s += __shfl_xor(s, 1);
            s += __shfl_xor(s, 2);
            s += __shfl_xor(s, 4);
            s += __shfl_xor(s, 8);
            lst[r] = lst[r] * scl[r] + s;
        }
#pragma unroll
        for (int nt = 0; nt < 4; ++nt)
#pragma unroll
            for (int r = 0; r < 8; ++r) acc[nt][r] *= scl[r];
        // P: C-layout f32 -> A-layout f16 via per-wave LDS (same-wave, DS in-order)
#pragma unroll
        for (int tt = 0; tt < 4; ++tt)
#pragma unroll
            for (int r = 0; r < 8; ++r)
                plds[w][r + 8 * half][tt * 16 + n16] = (_Float16)sc[tt][r];
        __asm__ volatile("s_wait_dscnt 0" ::: "memory");
#pragma unroll
        for (int kb2 = 0; kb2 < 64; kb2 += 32) {
            // cluster the 4 V-frag loads, then the A-frag, then 4 WMMAs
            v16h vf[4];
#pragma unroll
            for (int nt = 0; nt < 4; ++nt)
                vf[nt] = ldB(Vbase + (size_t)(nt * 16 + n16) * SS + tb, kb2, half);
            v16h pa = ldA(&plds[w][n16][0], kb2, half);
#pragma unroll
            for (int nt = 0; nt < 4; ++nt) acc[nt] = wmma32(pa, vf[nt], acc[nt]);
        }
    }

    // normalize (lst >= 1 always: each row contributed exp(0)=1 at its max tile)
#pragma unroll
    for (int r = 0; r < 8; ++r) {
        const float inv = 1.0f / lst[r];
#pragma unroll
        for (int nt = 0; nt < 4; ++nt) acc[nt][r] *= inv;
    }

    // fused per-head output projection: out = ctx(16x64) x Wo^T(64x64) + bo
#pragma unroll
    for (int nt = 0; nt < 4; ++nt)
#pragma unroll
        for (int r = 0; r < 8; ++r)
            plds[w][r + 8 * half][nt * 16 + n16] = (_Float16)acc[nt][r];
    __asm__ volatile("s_wait_dscnt 0" ::: "memory");

    v8f outv[4];
#pragma unroll
    for (int nt = 0; nt < 4; ++nt)
#pragma unroll
        for (int r = 0; r < 8; ++r) outv[nt][r] = 0.f;
#pragma unroll
    for (int kb = 0; kb < 64; kb += 32) {
        v16h wf[4];
#pragma unroll
        for (int nt = 0; nt < 4; ++nt)
            wf[nt] = ldB(WoT + (size_t)(h * 64 + nt * 16 + n16) * 64, kb, half);
        v16h ca = ldA(&plds[w][n16][0], kb, half);
#pragma unroll
        for (int nt = 0; nt < 4; ++nt) outv[nt] = wmma32(ca, wf[nt], outv[nt]);
    }
#pragma unroll
    for (int nt = 0; nt < 4; ++nt) {
        const int n = nt * 16 + n16;
        const float bval = bo[h * 64 + n];
#pragma unroll
        for (int r = 0; r < 8; ++r) {
            int m = r + 8 * half;
            xout[((size_t)b * SS + s0 + w * 16 + m) * DD + h * 64 + n] = outv[nt][r] + bval;
        }
    }
}

extern "C" void kernel_launch(void* const* d_in, const int* in_sizes, int n_in,
                              void* d_out, int out_size, void* d_ws, size_t ws_size,
                              hipStream_t stream) {
    const float* x_in = (const float*)d_in[0];
    const float* Wq = (const float*)d_in[1];
    const float* bq = (const float*)d_in[2];
    const float* Wk = (const float*)d_in[3];
    const float* bk = (const float*)d_in[4];
    const float* Wv = (const float*)d_in[5];
    const float* bv = (const float*)d_in[6];
    const float* Wo = (const float*)d_in[7];
    const float* bo = (const float*)d_in[8];

    // workspace carve (~64 MB total)
    char* p = (char*)d_ws;
    float* X = (float*)p;            p += (size_t)NB * SS * DD * 4;        // 25.2 MB fp32 activations
    _Float16* Qg = (_Float16*)p;     p += (size_t)NB * HH * SS * HD * 2;   // 12.6 MB
    _Float16* Kg = (_Float16*)p;     p += (size_t)NB * HH * SS * HD * 2;
    _Float16* Vt = (_Float16*)p;     p += (size_t)NB * HH * HD * SS * 2;   // transposed [hd][s]
    _Float16* WTq = (_Float16*)p;    p += (size_t)HH * HD * DD * 2;        // 1.18 MB each
    _Float16* WTk = (_Float16*)p;    p += (size_t)HH * HD * DD * 2;
    _Float16* WTv = (_Float16*)p;    p += (size_t)HH * HD * DD * 2;
    _Float16* WoT = (_Float16*)p;    p += (size_t)HH * HD * HD * 2;

    // one-time weight transpose/convert
    {
        int tot = HH * 64 * DD;
        cvt_wT_kernel<<<(tot + 255) / 256, 256, 0, stream>>>(Wq, WTq, DD);
        cvt_wT_kernel<<<(tot + 255) / 256, 256, 0, stream>>>(Wk, WTk, DD);
        cvt_wT_kernel<<<(tot + 255) / 256, 256, 0, stream>>>(Wv, WTv, DD);
        int toto = HH * 64 * 64;
        cvt_wT_kernel<<<(toto + 255) / 256, 256, 0, stream>>>(Wo, WoT, 64);
    }

    const int grid = NB * HH * (SS / 64);  // 1536
    for (int layer = 0; layer < 12; ++layer) {
        const float* xin = (layer == 0) ? x_in : X;
        float* xo = (layer == 11) ? (float*)d_out : X;
        qkv_proj_kernel<<<grid, 384, 0, stream>>>(xin, WTq, WTk, WTv, bq, bk, bv, Qg, Kg, Vt);
        attn_kernel<<<grid, 128, 0, stream>>>(Qg, Kg, Vt, WoT, bo, xo);
    }
}
